// Transformer_pointdecoder_base_v2_63840393887958
// MI455X (gfx1250) — matrix operand order repaired and checked by reading, not compiled, not run on hardware
//
#include <hip/hip_runtime.h>

// B=16, D=512, N=2048, K=32
constexpr int B_ = 16, D_ = 512, N_ = 2048, K_ = 32;
constexpr int NK = N_ * K_;   // 65536

typedef __attribute__((ext_vector_type(16))) __bf16       v16bf;
typedef __attribute__((ext_vector_type(8)))  float        v8f;
typedef __attribute__((ext_vector_type(4))) unsigned int  v4u;
typedef __attribute__((ext_vector_type(8)))  int          v8i;
typedef __attribute__((ext_vector_type(4)))  int          v4i;

union UV16 { v16bf v; struct { v4u lo, hi; } u; };

// LDS byte offset of a __shared__ pointer: generic (flat) address low 32 bits
// are the LDS offset per the aperture mapping (ISA 10.2).
__device__ __forceinline__ unsigned lds_off_of(const void* p) {
    return (unsigned)(unsigned long long)p;
}

// Issue a 2-D TDM tile load (bf16 elements) per CDNA5 ISA 8.3/8.4:
//  group0: count=1 | lds_addr | global_addr | type=2
//  group1: data_size=1(2B) | tensor_dim0/1 | tile_dim0/1 | tensor_dim0_stride
// This toolchain exposes the 6-arg builtin (groups 2/3 + extra group zeroed
// for a 2-D tile; cpol=0).
__device__ __forceinline__ void tdm_load_2d_bf16(unsigned lds_off, const void* gptr,
                                                 unsigned tensor_d0, unsigned tensor_d1,
                                                 unsigned tile_d0, unsigned tile_d1,
                                                 unsigned long long stride0)
{
    const unsigned long long ga = (unsigned long long)gptr;
    v4u g0;
    g0[0] = 1u;                                            // count=1, user D#
    g0[1] = lds_off;                                       // lds_addr (bytes)
    g0[2] = (unsigned)(ga & 0xffffffffu);                  // global_addr[31:0]
    g0[3] = (unsigned)((ga >> 32) & 0x1ffffffu)            // global_addr[56:32]
          | (2u << 30);                                    // type=2 ("image")
    v8i g1;
    g1[0] = (int)(1u << 16);                               // data_size=1 -> 2 bytes
    g1[1] = (int)((tensor_d0 & 0xffffu) << 16);            // tensor_dim0[15:0] @bit48
    g1[2] = (int)((tensor_d0 >> 16) |
                  ((tensor_d1 & 0xffffu) << 16));          // td0[31:16] | td1[15:0]
    g1[3] = (int)((tensor_d1 >> 16) | (tile_d0 << 16));    // td1[31:16] | tile_dim0
    g1[4] = (int)(tile_d1);                                // tile_dim1 (tile_dim2=0)
    g1[5] = (int)(stride0 & 0xffffffffu);                  // dim0_stride[31:0]
    g1[6] = (int)((stride0 >> 32) & 0xffffu);              // dim0_stride[47:32]
    g1[7] = 0;                                             // dim1_stride unused (2D)
    const v4i z4 = {0, 0, 0, 0};                           // groups 2/3 unused (2D)
    const v8i z8 = {0, 0, 0, 0, 0, 0, 0, 0};
    __builtin_amdgcn_tensor_load_to_lds(g0, g1, z4, z4, z8, 0);
}

// ---------------------------------------------------------------------------
// Kernel 1: qkv projection. C[16,65536] = x[16,512] @ W[512,65536] + b, x3.
// Memory-bound (384MB fp32 weight stream ~16.5us @ 23.3TB/s); bf16 WMMA.
// ---------------------------------------------------------------------------
__global__ __launch_bounds__(256)
void qkv_gemm_kernel(const float* __restrict__ x,
                     const float* __restrict__ WQ, const float* __restrict__ bQ,
                     const float* __restrict__ WK, const float* __restrict__ bK,
                     const float* __restrict__ WV, const float* __restrict__ bV,
                     __bf16* __restrict__ qb, __bf16* __restrict__ kTb,
                     __bf16* __restrict__ vb)
{
    const int lane = threadIdx.x & 31;
    const int wave = threadIdx.x >> 5;
    const int blocksPerMat = NK / 128;              // 512
    const int mat  = blockIdx.x / blocksPerMat;     // 0:Q 1:K 2:V
    const int tile = blockIdx.x % blocksPerMat;
    const int c0   = tile * 128 + wave * 16;        // 16 output columns per wave

    const float* W; const float* bias; __bf16* dstbase;
    if (mat == 0)      { W = WQ; bias = bQ; dstbase = qb;  }
    else if (mat == 1) { W = WK; bias = bK; dstbase = kTb; }
    else               { W = WV; bias = bV; dstbase = vb;  }

    const int arow = lane & 15;
    const int aklo = (lane >> 4) << 3;   // 0 or 8
    const int bcol = lane & 15;
    const int bklo = (lane >> 4) << 4;   // 0 or 16

    const float* xrow = x + arow * D_;
    v8f acc = {};

    for (int kk = 0; kk < D_; kk += 32) {
        v16bf a, b;
        #pragma unroll
        for (int j = 0; j < 8; ++j) {
            a[j]     = (__bf16)xrow[kk + 2 * aklo + j];
            a[8 + j] = (__bf16)xrow[kk + 2 * aklo + 8 + j];
        }
        const float* wc = W + (size_t)(kk + bklo) * NK + (c0 + bcol);
        #pragma unroll
        for (int j = 0; j < 16; ++j)
            b[j] = (__bf16)wc[(size_t)j * NK];
        if (kk + 32 < D_)
            __builtin_prefetch(wc + (size_t)32 * NK, 0, 3);
        acc = __builtin_amdgcn_wmma_f32_16x16x32_bf16(false, a, false, b,
                                                      (short)0, acc, false, false);
    }

    // C/D layout: acc[r] at row (r + 8*(lane>=16)), col c0 + (lane&15)
    const int col  = c0 + (lane & 15);
    const float bvl = bias[col];
    const int n  = col >> 5;
    const int kd = col & 31;
    // Q,V stored [b,n,32]; K stored transposed [b,32,n] (branch-free offset)
    const size_t colOff = (mat == 1) ? ((size_t)kd * N_ + n)
                                     : ((size_t)n * K_ + kd);
    const int mbase = (lane < 16) ? 0 : 8;
    #pragma unroll
    for (int r = 0; r < 8; ++r)
        dstbase[(size_t)(r + mbase) * NK + colOff] = (__bf16)(acc[r] + bvl);
}

// ---------------------------------------------------------------------------
// Kernel 2: flash attention + fused 3-channel out/ori projections.
// K/V tiles are DMA'd once per block by the Tensor Data Mover into LDS
// (double-buffered), shared by all 8 waves; never materializes 2048x2048.
// ---------------------------------------------------------------------------
__global__ __launch_bounds__(256)
void attn_fused_kernel(const __bf16* __restrict__ qb, const __bf16* __restrict__ kTb,
                       const __bf16* __restrict__ vb,
                       const float* __restrict__ Wout, const float* __restrict__ bout,
                       const float* __restrict__ Wori, const float* __restrict__ bori,
                       float* __restrict__ out)
{
    __shared__ __align__(16) __bf16 ldsKT[2][32][64];   // kT tile: [kd][m]
    __shared__ __align__(16) __bf16 ldsV [2][64][32];   // V tile:  [m][kd]
    __shared__ __align__(16) __bf16 plds[8][16][64];    // per-wave P staging
    __shared__ float orilds[8][16][3];

    const int lane = threadIdx.x & 31;
    const int wave = threadIdx.x >> 5;
    const int b  = blockIdx.x >> 4;                 // batch
    const int qt = (blockIdx.x & 15) * 8 + wave;    // q tile 0..127
    const int n0 = qt * 16;

    const int arow = lane & 15;
    const int aklo = (lane >> 4) << 3;   // 0/8
    const int bcol = lane & 15;
    const int bklo = (lane >> 4) << 4;   // 0/16
    const int g    = lane >> 4;

    // q A-tile: 16 rows x 32 K, kept in registers for all KV blocks
    v16bf aq;
    {
        const __bf16* qr = qb + ((size_t)b * N_ + n0 + arow) * K_;
        UV16 t;
        t.u.lo = *(const v4u*)(qr + aklo);
        t.u.hi = *(const v4u*)(qr + 16 + aklo);
        aq = t.v;
    }

    const float SCALE = 0.17677669529663687f;   // 1/sqrt(K)

    v8f o0 = {}, o1 = {};
    float rowmax[8], rowsum[8];
    #pragma unroll
    for (int r = 0; r < 8; ++r) { rowmax[r] = -1e30f; rowsum[r] = 0.f; }

    // TDM issue: one wave DMAs the 64-row K/V tiles for m-block `mb` into buf
    auto issue_block = [&](int mb, int buf) {
        const __bf16* kg = kTb + (size_t)b * NK + mb * 64;          // kT[b][0][m0]
        tdm_load_2d_bf16(lds_off_of(&ldsKT[buf][0][0]), kg,
                         /*tensor*/ N_, K_, /*tile*/ 64, 32, /*stride0*/ N_);
        const __bf16* vg = vb + ((size_t)b * N_ + (size_t)mb * 64) * K_;
        tdm_load_2d_bf16(lds_off_of(&ldsV[buf][0][0]), vg,
                         /*tensor*/ K_, N_, /*tile*/ 32, 64, /*stride0*/ K_);
    };

    constexpr int NBLK = N_ / 64;   // 32
    if (wave == 0) issue_block(0, 0);

    for (int i = 0; i < NBLK; ++i) {
        const int buf = i & 1;
        if (wave == 0) {
            if (i + 1 < NBLK) {
                issue_block(i + 1, (i + 1) & 1);
                __builtin_amdgcn_s_wait_tensorcnt(2);   // block i's 2 DMAs done
            } else {
                __builtin_amdgcn_s_wait_tensorcnt(0);
            }
        }
        __syncthreads();    // LDS tiles for block i visible to all waves

        // S = q @ kT : 4 WMMAs -> 16x64 fp32 tile in VGPRs
        v8f s[4];
        #pragma unroll
        for (int t = 0; t < 4; ++t) {
            v16bf bk;
            const __bf16* kp = &ldsKT[buf][aklo][t * 16 + bcol];
            #pragma unroll
            for (int j = 0; j < 16; ++j) bk[j] = kp[(j < 8 ? j : j + 8) * 64];
            v8f z = {};
            s[t] = __builtin_amdgcn_wmma_f32_16x16x32_bf16(false, aq, false, bk,
                                                           (short)0, z, false, false);
        }

        // online softmax: per-row block max (reduce over the 16-lane group)
        #pragma unroll
        for (int r = 0; r < 8; ++r) {
            float v = fmaxf(fmaxf(s[0][r], s[1][r]), fmaxf(s[2][r], s[3][r]));
            v = fmaxf(v, __shfl_xor(v, 1));
            v = fmaxf(v, __shfl_xor(v, 2));
            v = fmaxf(v, __shfl_xor(v, 4));
            v = fmaxf(v, __shfl_xor(v, 8));
            const float mnew  = fmaxf(rowmax[r], v * SCALE);
            const float alpha = __expf(rowmax[r] - mnew);
            rowmax[r] = mnew;
            rowsum[r] *= alpha;
            o0[r] *= alpha;
            o1[r] *= alpha;
        }

        // P = exp(S*scale - rowmax), staged D-layout -> LDS (same-wave DS in-order)
        #pragma unroll
        for (int t = 0; t < 4; ++t) {
            #pragma unroll
            for (int r = 0; r < 8; ++r) {
                const float p = __expf(s[t][r] * SCALE - rowmax[r]);
                rowsum[r] += p;
                plds[wave][r + 8 * g][t * 16 + bcol] = (__bf16)p;
            }
        }
        __builtin_amdgcn_wave_barrier();

        // O += P @ V : 2 m-chunks x 2 kd tiles = 4 WMMAs
        #pragma unroll
        for (int ch = 0; ch < 2; ++ch) {
            v16bf pa;
            {
                const __bf16* pr = &plds[wave][arow][ch * 32];
                UV16 t;
                t.u.lo = *(const v4u*)(pr + aklo);
                t.u.hi = *(const v4u*)(pr + 16 + aklo);
                pa = t.v;
            }
            #pragma unroll
            for (int kt = 0; kt < 2; ++kt) {
                v16bf bvv;
                const __bf16* vp = &ldsV[buf][ch * 32 + aklo][kt * 16 + bcol];
                #pragma unroll
                for (int j = 0; j < 16; ++j) bvv[j] = vp[(j < 8 ? j : j + 8) * 32];
                if (kt == 0)
                    o0 = __builtin_amdgcn_wmma_f32_16x16x32_bf16(false, pa, false, bvv,
                                                                 (short)0, o0, false, false);
                else
                    o1 = __builtin_amdgcn_wmma_f32_16x16x32_bf16(false, pa, false, bvv,
                                                                 (short)0, o1, false, false);
            }
        }
        __syncthreads();    // all waves done reading buf before it is refilled
    }

    // finalize softmax denominators and normalize O
    #pragma unroll
    for (int r = 0; r < 8; ++r) {
        float v = rowsum[r];
        v += __shfl_xor(v, 1); v += __shfl_xor(v, 2);
        v += __shfl_xor(v, 4); v += __shfl_xor(v, 8);
        const float inv = __builtin_amdgcn_rcpf(v);
        o0[r] *= inv;
        o1[r] *= inv;
    }

    // ori = q @ Wori^T + bori  (A-layout: lane pair (L, L^16) shares row L&15)
    #pragma unroll
    for (int c = 0; c < 3; ++c) {
        float acc = 0.f;
        #pragma unroll
        for (int j = 0; j < 8; ++j) {
            acc += (float)aq[j]     * Wori[c * K_ + aklo + j];
            acc += (float)aq[8 + j] * Wori[c * K_ + 16 + aklo + j];
        }
        acc += __shfl_xor(acc, 16);
        if (lane < 16) orilds[wave][lane][c] = acc + bori[c];
    }
    __builtin_amdgcn_wave_barrier();

    // out3 = O @ Wout^T + bout + ori  (D-layout: reduce over 16-lane group)
    #pragma unroll
    for (int c = 0; c < 3; ++c) {
        const float w0 = Wout[c * K_ + bcol];
        const float w1 = Wout[c * K_ + 16 + bcol];
        #pragma unroll
        for (int r = 0; r < 8; ++r) {
            float v = o0[r] * w0 + o1[r] * w1;
            v += __shfl_xor(v, 1); v += __shfl_xor(v, 2);
            v += __shfl_xor(v, 4); v += __shfl_xor(v, 8);
            if (bcol == 0) {
                const int row = r + 8 * g;
                out[((size_t)b * N_ + n0 + row) * 3 + c] =
                    v + bout[c] + orilds[wave][row][c];
            }
        }
    }
}

// ---------------------------------------------------------------------------
extern "C" void kernel_launch(void* const* d_in, const int* in_sizes, int n_in,
                              void* d_out, int out_size, void* d_ws, size_t ws_size,
                              hipStream_t stream) {
    const float* x    = (const float*)d_in[0];
    const float* WQ   = (const float*)d_in[1];
    const float* bQ   = (const float*)d_in[2];
    const float* WK   = (const float*)d_in[3];
    const float* bK   = (const float*)d_in[4];
    const float* WV   = (const float*)d_in[5];
    const float* bV   = (const float*)d_in[6];
    const float* Wout = (const float*)d_in[7];
    const float* bout = (const float*)d_in[8];
    const float* Wori = (const float*)d_in[9];
    const float* bori = (const float*)d_in[10];
    float* out = (float*)d_out;

    // workspace: q[16,2048,32] bf16, kT[16,32,2048] bf16, v[16,2048,32] bf16 = 6MB
    __bf16* qb  = (__bf16*)d_ws;
    __bf16* kTb = qb  + (size_t)B_ * NK;
    __bf16* vb  = kTb + (size_t)B_ * NK;

    qkv_gemm_kernel<<<3 * (NK / 128), 256, 0, stream>>>(
        x, WQ, bQ, WK, bK, WV, bV, qb, kTb, vb);
    attn_fused_kernel<<<B_ * (N_ / 16 / 8), 256, 0, stream>>>(
        qb, kTb, vb, Wout, bout, Wori, bori, out);
}
